// PipelinedGemm_72516227825730
// MI455X (gfx1250) — matrix operand order repaired and checked by reading, not compiled, not run on hardware
//
#include <hip/hip_runtime.h>

typedef float v8f __attribute__((ext_vector_type(8)));
typedef __bf16 bf16_t;
typedef bf16_t v4bf  __attribute__((ext_vector_type(4)));
typedef bf16_t v8bf  __attribute__((ext_vector_type(8)));
typedef bf16_t v16bf __attribute__((ext_vector_type(16)));

union BFrag {
  v16bf v;
  v8bf  h[2];
};

#define BM 128
#define BN 128
#define BKF 32   // fused fallback path
#define BKP 64   // pre-split fast path
#define EPILOGUE_CONST 36.0f  // sum_{r=0}^{7} (r+1)

// ---------------- async global->LDS staging (gfx1250 ASYNCcnt path) --------
#if defined(__HIP_DEVICE_COMPILE__) &&                                        \
    __has_builtin(__builtin_amdgcn_global_load_async_to_lds_b128) &&          \
    __has_builtin(__builtin_amdgcn_s_wait_asynccnt)
#define USE_ASYNC_LDS 1
#else
#define USE_ASYNC_LDS 0
#endif

typedef int v4i_t __attribute__((vector_size(16)));
#if USE_ASYNC_LDS
typedef __attribute__((address_space(1))) v4i_t* as1_v4i_ptr;  // global
typedef __attribute__((address_space(3))) v4i_t* as3_v4i_ptr;  // LDS
#endif

__device__ __forceinline__ void copy16_g2l(const bf16_t* g, bf16_t* l) {
#if USE_ASYNC_LDS
  __builtin_amdgcn_global_load_async_to_lds_b128(
      (as1_v4i_ptr)(g), (as3_v4i_ptr)(l), 0, 0);
#else
  *(v8bf*)l = *(const v8bf*)g;
#endif
}

__device__ __forceinline__ void async_join() {
#if USE_ASYNC_LDS
  __builtin_amdgcn_s_wait_asynccnt(0);
#endif
}

// ---------------- pre-pass 1: elementwise split of A -----------------------
__global__ __launch_bounds__(256)
void split_a_kernel(const float* __restrict__ A, bf16_t* __restrict__ Ahi,
                    bf16_t* __restrict__ Alo) {
  const size_t idx = ((size_t)blockIdx.x * 256 + threadIdx.x) * 8;
#pragma unroll
  for (int half = 0; half < 2; ++half) {
    const float4 v = *(const float4*)(A + idx + half * 4);
    const float xs[4] = {v.x, v.y, v.z, v.w};
    v4bf hi, lo;
#pragma unroll
    for (int j = 0; j < 4; ++j) {
      const bf16_t h = (bf16_t)xs[j];  // RNE high part
      hi[j] = h;
      lo[j] = (bf16_t)(xs[j] - (float)h);  // residual
    }
    *(v4bf*)(Ahi + idx + half * 4) = hi;
    *(v4bf*)(Alo + idx + half * 4) = lo;
  }
}

// ---------------- pre-pass 2: split + transpose B (K x N -> N x K) ---------
#define TT 32
__global__ __launch_bounds__(256)
void split_bt_kernel(const float* __restrict__ B, bf16_t* __restrict__ Bthi,
                     bf16_t* __restrict__ Btlo, int N, int K) {
  __shared__ __align__(16) bf16_t Lhi[TT * TT];
  __shared__ __align__(16) bf16_t Llo[TT * TT];
  const int k0 = blockIdx.y * TT;  // K block
  const int n0 = blockIdx.x * TT;  // N block
  const int t  = threadIdx.x;
  const int r  = t >> 3;         // 0..31
  const int c4 = (t & 7) << 2;   // 0..28

  const float4 v = *(const float4*)(B + (size_t)(k0 + r) * N + (n0 + c4));
  const float xs[4] = {v.x, v.y, v.z, v.w};
#pragma unroll
  for (int j = 0; j < 4; ++j) {
    const bf16_t h = (bf16_t)xs[j];
    Lhi[(c4 + j) * TT + r] = h;               // transposed LDS store
    Llo[(c4 + j) * TT + r] = (bf16_t)(xs[j] - (float)h);
  }
  __syncthreads();
  const v4bf oh = *(const v4bf*)(&Lhi[r * TT + c4]);
  const v4bf ol = *(const v4bf*)(&Llo[r * TT + c4]);
  *(v4bf*)(Bthi + (size_t)(n0 + r) * K + (k0 + c4)) = oh;  // coalesced out
  *(v4bf*)(Btlo + (size_t)(n0 + r) * K + (k0 + c4)) = ol;
}

// ---------------- main GEMM, pre-split fast path ---------------------------
__global__ __launch_bounds__(256)
void gemm_wmma_bf16x3_presplit(const bf16_t* __restrict__ Ahi,
                               const bf16_t* __restrict__ Alo,
                               const bf16_t* __restrict__ Bthi,
                               const bf16_t* __restrict__ Btlo,
                               float* __restrict__ C, int M, int N, int K) {
  __shared__ __align__(16) bf16_t AsHi[BM * BKP];
  __shared__ __align__(16) bf16_t AsLo[BM * BKP];
  __shared__ __align__(16) bf16_t BsHi[BN * BKP];  // n-major (pre-transposed)
  __shared__ __align__(16) bf16_t BsLo[BN * BKP];

  const int tid   = threadIdx.x;
  const int lane  = tid & 31;
  const int wave  = tid >> 5;
  const int waveM = wave & 3;
  const int waveN = wave >> 2;
  const int blockM = blockIdx.y * BM;
  const int blockN = blockIdx.x * BN;
  const int lhalf = lane >> 4;
  const int l16   = lane & 15;

  v8f acc[2][4];
#pragma unroll
  for (int mt = 0; mt < 2; ++mt)
#pragma unroll
    for (int nt = 0; nt < 4; ++nt)
#pragma unroll
      for (int e = 0; e < 8; ++e)
        acc[mt][nt][e] = 0.0f;

  for (int k0 = 0; k0 < K; k0 += BKP) {
    // ---- Stage 4x (128x64 bf16) tiles: pure contiguous 16B copies ----
#pragma unroll
    for (int i = 0; i < 4; ++i) {
      const int f  = tid + i * 256;   // 0..1023
      const int r  = f >> 3;          // 0..127
      const int c8 = (f & 7) << 3;    // 0..56
      const size_t ga = (size_t)(blockM + r) * K + (k0 + c8);
      const size_t gb = (size_t)(blockN + r) * K + (k0 + c8);
      copy16_g2l(Ahi + ga, &AsHi[r * BKP + c8]);
      copy16_g2l(Alo + ga, &AsLo[r * BKP + c8]);
      copy16_g2l(Bthi + gb, &BsHi[r * BKP + c8]);
      copy16_g2l(Btlo + gb, &BsLo[r * BKP + c8]);
    }

    if (k0 + BKP < K) {
      __builtin_prefetch(Ahi + (size_t)(blockM + (tid >> 1)) * K + (k0 + BKP), 0, 1);
      __builtin_prefetch(Bthi + (size_t)(blockN + (tid >> 1)) * K + (k0 + BKP), 0, 1);
    }

    async_join();
    __syncthreads();

#pragma unroll
    for (int ks = 0; ks < BKP / 32; ++ks) {
      // A 16x32 bf16 layout: lanes 0-15 K=0..7 & 16..23; lanes 16-31 K=8..15 & 24..31
      BFrag aHi[2], aLo[2], bHi[4], bLo[4];
#pragma unroll
      for (int mt = 0; mt < 2; ++mt) {
        const int row  = waveM * 32 + mt * 16 + l16;
        const int base = row * BKP + ks * 32 + (lhalf << 3);
        aHi[mt].h[0] = *(const v8bf*)(&AsHi[base]);
        aHi[mt].h[1] = *(const v8bf*)(&AsHi[base + 16]);
        aLo[mt].h[0] = *(const v8bf*)(&AsLo[base]);
        aLo[mt].h[1] = *(const v8bf*)(&AsLo[base + 16]);
      }
      // B 32x16 bf16 layout: lanes 0-15 K=0..7 & 16..23; lanes 16-31 K=8..15 & 24..31
#pragma unroll
      for (int nt = 0; nt < 4; ++nt) {
        const int col  = waveN * 64 + nt * 16 + l16;
        const int base = col * BKP + ks * 32 + (lhalf << 3);
        bHi[nt].h[0] = *(const v8bf*)(&BsHi[base]);
        bHi[nt].h[1] = *(const v8bf*)(&BsHi[base + 16]);
        bLo[nt].h[0] = *(const v8bf*)(&BsLo[base]);
        bLo[nt].h[1] = *(const v8bf*)(&BsLo[base + 16]);
      }
#pragma unroll
      for (int mt = 0; mt < 2; ++mt) {
#pragma unroll
        for (int nt = 0; nt < 4; ++nt) {
          acc[mt][nt] = __builtin_amdgcn_wmma_f32_16x16x32_bf16(
              false, aHi[mt].v, false, bHi[nt].v, (short)0, acc[mt][nt],
              false, false);
          acc[mt][nt] = __builtin_amdgcn_wmma_f32_16x16x32_bf16(
              false, aHi[mt].v, false, bLo[nt].v, (short)0, acc[mt][nt],
              false, false);
          acc[mt][nt] = __builtin_amdgcn_wmma_f32_16x16x32_bf16(
              false, aLo[mt].v, false, bHi[nt].v, (short)0, acc[mt][nt],
              false, false);
        }
      }
    }

    __syncthreads();
  }

  const int mBase = blockM + waveM * 32;
  const int nBase = blockN + waveN * 64;
#pragma unroll
  for (int mt = 0; mt < 2; ++mt) {
#pragma unroll
    for (int nt = 0; nt < 4; ++nt) {
      const int n  = nBase + nt * 16 + l16;
      const int m0 = mBase + mt * 16 + lhalf * 8;
#pragma unroll
      for (int v = 0; v < 8; ++v) {
        C[(size_t)(m0 + v) * N + n] = acc[mt][nt][v] + EPILOGUE_CONST;
      }
    }
  }
}

// ---------------- fused fallback (no workspace needed) ---------------------
__global__ __launch_bounds__(256)
void gemm_wmma_bf16x3_fused(const float* __restrict__ A,
                            const float* __restrict__ B,
                            float* __restrict__ C, int M, int N, int K) {
  __shared__ __align__(16) bf16_t AsHi[BM * BKF];
  __shared__ __align__(16) bf16_t AsLo[BM * BKF];
  __shared__ __align__(16) bf16_t BsHi[BN * BKF];
  __shared__ __align__(16) bf16_t BsLo[BN * BKF];

  const int tid   = threadIdx.x;
  const int lane  = tid & 31;
  const int wave  = tid >> 5;
  const int waveM = wave & 3;
  const int waveN = wave >> 2;
  const int blockM = blockIdx.y * BM;
  const int blockN = blockIdx.x * BN;
  const int lhalf = lane >> 4;
  const int l16   = lane & 15;

  v8f acc[2][4];
#pragma unroll
  for (int mt = 0; mt < 2; ++mt)
#pragma unroll
    for (int nt = 0; nt < 4; ++nt)
#pragma unroll
      for (int e = 0; e < 8; ++e)
        acc[mt][nt][e] = 0.0f;

  for (int k0 = 0; k0 < K; k0 += BKF) {
#pragma unroll
    for (int i = 0; i < 4; ++i) {
      const int f = tid + i * 256;
      const int r = f >> 3;
      const int c = (f & 7) << 2;
      const float4 va = *(const float4*)(A + (size_t)(blockM + r) * K + (k0 + c));
      const float xs[4] = {va.x, va.y, va.z, va.w};
      v4bf hi, lo;
#pragma unroll
      for (int j = 0; j < 4; ++j) {
        const bf16_t h = (bf16_t)xs[j];
        hi[j] = h;
        lo[j] = (bf16_t)(xs[j] - (float)h);
      }
      *(v4bf*)(&AsHi[r * BKF + c]) = hi;
      *(v4bf*)(&AsLo[r * BKF + c]) = lo;
    }
#pragma unroll
    for (int i = 0; i < 4; ++i) {
      const int f  = tid + i * 256;
      const int kr = f >> 5;
      const int cc = (f & 31) << 2;
      const float4 vb = *(const float4*)(B + (size_t)(k0 + kr) * N + (blockN + cc));
      const float xs[4] = {vb.x, vb.y, vb.z, vb.w};
#pragma unroll
      for (int j = 0; j < 4; ++j) {
        const bf16_t h = (bf16_t)xs[j];
        BsHi[(cc + j) * BKF + kr] = h;
        BsLo[(cc + j) * BKF + kr] = (bf16_t)(xs[j] - (float)h);
      }
    }
    __syncthreads();

    BFrag aHi[2], aLo[2], bHi[4], bLo[4];
#pragma unroll
    for (int mt = 0; mt < 2; ++mt) {
      const int row  = waveM * 32 + mt * 16 + l16;
      const int base = row * BKF + (lhalf << 3);
      aHi[mt].h[0] = *(const v8bf*)(&AsHi[base]);
      aHi[mt].h[1] = *(const v8bf*)(&AsHi[base + 16]);
      aLo[mt].h[0] = *(const v8bf*)(&AsLo[base]);
      aLo[mt].h[1] = *(const v8bf*)(&AsLo[base + 16]);
    }
#pragma unroll
    for (int nt = 0; nt < 4; ++nt) {
      const int col  = waveN * 64 + nt * 16 + l16;
      const int base = col * BKF + (lhalf << 3);
      bHi[nt].h[0] = *(const v8bf*)(&BsHi[base]);
      bHi[nt].h[1] = *(const v8bf*)(&BsHi[base + 16]);
      bLo[nt].h[0] = *(const v8bf*)(&BsLo[base]);
      bLo[nt].h[1] = *(const v8bf*)(&BsLo[base + 16]);
    }
#pragma unroll
    for (int mt = 0; mt < 2; ++mt) {
#pragma unroll
      for (int nt = 0; nt < 4; ++nt) {
        acc[mt][nt] = __builtin_amdgcn_wmma_f32_16x16x32_bf16(
            false, aHi[mt].v, false, bHi[nt].v, (short)0, acc[mt][nt], false, false);
        acc[mt][nt] = __builtin_amdgcn_wmma_f32_16x16x32_bf16(
            false, aHi[mt].v, false, bLo[nt].v, (short)0, acc[mt][nt], false, false);
        acc[mt][nt] = __builtin_amdgcn_wmma_f32_16x16x32_bf16(
            false, aLo[mt].v, false, bHi[nt].v, (short)0, acc[mt][nt], false, false);
      }
    }
    __syncthreads();
  }

  const int mBase = blockM + waveM * 32;
  const int nBase = blockN + waveN * 64;
#pragma unroll
  for (int mt = 0; mt < 2; ++mt) {
#pragma unroll
    for (int nt = 0; nt < 4; ++nt) {
      const int n  = nBase + nt * 16 + l16;
      const int m0 = mBase + mt * 16 + lhalf * 8;
#pragma unroll
      for (int v = 0; v < 8; ++v) {
        C[(size_t)(m0 + v) * N + n] = acc[mt][nt][v] + EPILOGUE_CONST;
      }
    }
  }
}

extern "C" void kernel_launch(void* const* d_in, const int* in_sizes, int n_in,
                              void* d_out, int out_size, void* d_ws, size_t ws_size,
                              hipStream_t stream) {
  const float* A = (const float*)d_in[0];
  const float* B = (const float*)d_in[1];
  float* C = (float*)d_out;
  const int M = 4096, N = 4096, K = 4096;
  const size_t elems = (size_t)M * K;
  const size_t need  = 4 * elems * sizeof(bf16_t);  // 128 MB

  dim3 grid(N / BN, M / BM);
  dim3 block(256);

  if (ws_size >= need) {
    bf16_t* Ahi  = (bf16_t*)d_ws;
    bf16_t* Alo  = Ahi + elems;
    bf16_t* Bthi = Alo + elems;
    bf16_t* Btlo = Bthi + elems;
    hipLaunchKernelGGL(split_a_kernel, dim3(elems / (256 * 8)), block, 0,
                       stream, A, Ahi, Alo);
    hipLaunchKernelGGL(split_bt_kernel, dim3(N / TT, K / TT), block, 0, stream,
                       B, Bthi, Btlo, N, K);
    hipLaunchKernelGGL(gemm_wmma_bf16x3_presplit, grid, block, 0, stream, Ahi,
                       Alo, Bthi, Btlo, C, M, N, K);
  } else {
    hipLaunchKernelGGL(gemm_wmma_bf16x3_fused, grid, block, 0, stream, A, B, C,
                       M, N, K);
  }
}
